// BitLinear_63204738728221
// MI455X (gfx1250) — 8-bit matrix operand order repaired and checked by reading, compiled, not run on hardware
//
#include <hip/hip_runtime.h>

typedef __attribute__((ext_vector_type(8))) int v8i;

#define Q_EPS   1e-8f
#define QMAXF   127.0f

#define TILE_M     128
#define TILE_N     128
#define TILE_K     64
#define LDS_STRIDE 80
#define TILE_BYTES (128 * LDS_STRIDE)
#define BUF_BYTES  (2 * TILE_BYTES)

__global__ __launch_bounds__(256) void aquant_kernel(
    const float* __restrict__ X, signed char* __restrict__ Aq,
    float* __restrict__ ascale, int K)
{
    const int row = blockIdx.x;
    const float4* x4 = reinterpret_cast<const float4*>(X + (size_t)row * K);
    const int n4 = K >> 2;

    float mx = 0.0f;
    for (int i = threadIdx.x; i < n4; i += 256) {
        float4 v = x4[i];
        mx = fmaxf(mx, fmaxf(fmaxf(fabsf(v.x), fabsf(v.y)),
                             fmaxf(fabsf(v.z), fabsf(v.w))));
    }
    __shared__ float red[256];
    red[threadIdx.x] = mx;
    __syncthreads();
    for (int off = 128; off > 0; off >>= 1) {
        if (threadIdx.x < off)
            red[threadIdx.x] = fmaxf(red[threadIdx.x], red[threadIdx.x + off]);
        __syncthreads();
    }
    const float scale = red[0] + Q_EPS;
    if (threadIdx.x == 0) ascale[row] = scale;
    const float s = QMAXF / scale;

    int* q4 = reinterpret_cast<int*>(Aq + (size_t)row * K);
    for (int i = threadIdx.x; i < n4; i += 256) {
        float4 v = x4[i];
        int q0 = (int)rintf(fminf(127.0f, fmaxf(-128.0f, v.x * s)));
        int q1 = (int)rintf(fminf(127.0f, fmaxf(-128.0f, v.y * s)));
        int q2 = (int)rintf(fminf(127.0f, fmaxf(-128.0f, v.z * s)));
        int q3 = (int)rintf(fminf(127.0f, fmaxf(-128.0f, v.w * s)));
        q4[i] = (q0 & 255) | ((q1 & 255) << 8) | ((q2 & 255) << 16) | ((q3 & 255) << 24);
    }
}

__global__ __launch_bounds__(256) void wquant_kernel(
    const float* __restrict__ W, signed char* __restrict__ Wq,
    float* __restrict__ wscale, int K)
{
    const int row = blockIdx.x;
    const float4* w4 = reinterpret_cast<const float4*>(W + (size_t)row * K);
    const int n4 = K >> 2;

    float s = 0.0f;
    for (int i = threadIdx.x; i < n4; i += 256) {
        float4 v = w4[i];
        s += fabsf(v.x) + fabsf(v.y) + fabsf(v.z) + fabsf(v.w);
    }
    __shared__ float red[256];
    red[threadIdx.x] = s;
    __syncthreads();
    for (int off = 128; off > 0; off >>= 1) {
        if (threadIdx.x < off) red[threadIdx.x] += red[threadIdx.x + off];
        __syncthreads();
    }
    const float scale = red[0] / (float)K + Q_EPS;
    if (threadIdx.x == 0) wscale[row] = scale;
    const float inv = 1.0f / scale;

    int* q4 = reinterpret_cast<int*>(Wq + (size_t)row * K);
    for (int i = threadIdx.x; i < n4; i += 256) {
        float4 v = w4[i];
        int q0 = (int)fminf(1.0f, fmaxf(-1.0f, rintf(v.x * inv)));
        int q1 = (int)fminf(1.0f, fmaxf(-1.0f, rintf(v.y * inv)));
        int q2 = (int)fminf(1.0f, fmaxf(-1.0f, rintf(v.z * inv)));
        int q3 = (int)fminf(1.0f, fmaxf(-1.0f, rintf(v.w * inv)));
        q4[i] = (q0 & 255) | ((q1 & 255) << 8) | ((q2 & 255) << 16) | ((q3 & 255) << 24);
    }
}

__device__ __forceinline__ unsigned lds_off(const void* p) {
    return (unsigned)(size_t)p;
}

__device__ __forceinline__ void async_copy16(unsigned ldsoff, const signed char* g) {
    asm volatile("global_load_async_to_lds_b128 %0, %1, off"
                 :: "v"(ldsoff), "v"((unsigned long long)(size_t)g)
                 : "memory");
}

__device__ __forceinline__ void wait_asynccnt0() {
#if __has_builtin(__builtin_amdgcn_s_wait_asynccnt)
    __builtin_amdgcn_s_wait_asynccnt(0);
#else
    asm volatile("s_wait_asynccnt 0" ::: "memory");
#endif
}

__device__ __forceinline__ v8i load_a_frag(const signed char* p) {
    uint2 d0 = *reinterpret_cast<const uint2*>(p);
    uint2 d1 = *reinterpret_cast<const uint2*>(p + 16);
    uint2 d2 = *reinterpret_cast<const uint2*>(p + 32);
    uint2 d3 = *reinterpret_cast<const uint2*>(p + 48);
    v8i a;
    a[0] = (int)d0.x; a[1] = (int)d0.y;
    a[2] = (int)d1.x; a[3] = (int)d1.y;
    a[4] = (int)d2.x; a[5] = (int)d2.y;
    a[6] = (int)d3.x; a[7] = (int)d3.y;
    return a;
}

__device__ __forceinline__ v8i load_b_frag(const signed char* p) {
    uint4 d0 = *reinterpret_cast<const uint4*>(p);
    uint4 d1 = *reinterpret_cast<const uint4*>(p + 32);
    v8i b;
    b[0] = (int)d0.x; b[1] = (int)d0.y; b[2] = (int)d0.z; b[3] = (int)d0.w;
    b[4] = (int)d1.x; b[5] = (int)d1.y; b[6] = (int)d1.z; b[7] = (int)d1.w;
    return b;
}

__global__ __launch_bounds__(256) void bitgemm_kernel(
    const signed char* __restrict__ Aq, const signed char* __restrict__ Wq,
    const float* __restrict__ ascale, const float* __restrict__ wscale,
    const float* __restrict__ alpha, float* __restrict__ Y,
    int M, int N, int K)
{
    __shared__ __align__(16) signed char smem[2 * BUF_BYTES];

    const int t     = threadIdx.x;
    const int lane  = t & 31;
    const int wave  = t >> 5;
    const int waveM = wave >> 1;
    const int waveN = wave & 1;

    const int rowBlk = blockIdx.y * TILE_M;
    const int colBlk = blockIdx.x * TILE_N;
    const int row0   = rowBlk + waveM * 32;
    const int col0   = colBlk + waveN * 64;

    const int c16 = (t & 3) * 16;
    const int r0  = t >> 2;
    const signed char* aRow0 = Aq + (size_t)(rowBlk + r0) * K + c16;
    const signed char* aRow1 = aRow0 + (size_t)64 * K;
    const signed char* bRow0 = Wq + (size_t)(colBlk + r0) * K + c16;
    const signed char* bRow1 = bRow0 + (size_t)64 * K;
    const unsigned smemBase  = lds_off(smem);
    const unsigned ldsRow0   = (unsigned)(r0 * LDS_STRIDE + c16);
    const unsigned ldsRow1   = (unsigned)((r0 + 64) * LDS_STRIDE + c16);

    const int ml = lane & 15;
    const int aK = (lane >> 4) << 4;
    const int bK = (lane >> 4) << 4;
    const int aFragOff = (waveM * 32 + ml) * LDS_STRIDE + aK;
    const int bFragOff = TILE_BYTES + (waveN * 64 + ml) * LDS_STRIDE + bK;

    v8i acc[2][4] = {};

    const int KT = K / TILE_K;

    {
        const unsigned aBuf = smemBase;
        const unsigned bBuf = smemBase + TILE_BYTES;
        async_copy16(aBuf + ldsRow0, aRow0);
        async_copy16(aBuf + ldsRow1, aRow1);
        async_copy16(bBuf + ldsRow0, bRow0);
        async_copy16(bBuf + ldsRow1, bRow1);
    }
    wait_asynccnt0();
    __syncthreads();

    for (int kt = 0; kt < KT; ++kt) {
        const int buf = kt & 1;

        if (kt + 1 < KT) {
            const int knext = (kt + 1) * TILE_K;
            const unsigned aBuf = smemBase + (unsigned)((buf ^ 1) * BUF_BYTES);
            const unsigned bBuf = aBuf + TILE_BYTES;
            async_copy16(aBuf + ldsRow0, aRow0 + knext);
            async_copy16(aBuf + ldsRow1, aRow1 + knext);
            async_copy16(bBuf + ldsRow0, bRow0 + knext);
            async_copy16(bBuf + ldsRow1, bRow1 + knext);
        }

        const signed char* base = smem + buf * BUF_BYTES;
        const signed char* aB = base + aFragOff;
        const signed char* bB = base + bFragOff;

        v8i af0 = load_b_frag(aB);
        v8i af1 = load_b_frag(aB + 16 * LDS_STRIDE);
        v8i bf0 = load_b_frag(bB);
        v8i bf1 = load_b_frag(bB + 16 * LDS_STRIDE);
        v8i bf2 = load_b_frag(bB + 32 * LDS_STRIDE);
        v8i bf3 = load_b_frag(bB + 48 * LDS_STRIDE);

        acc[0][0] = __builtin_amdgcn_wmma_i32_16x16x64_iu8(true, af0, true, bf0, acc[0][0], false, false);
        acc[1][0] = __builtin_amdgcn_wmma_i32_16x16x64_iu8(true, af1, true, bf0, acc[1][0], false, false);
        acc[0][1] = __builtin_amdgcn_wmma_i32_16x16x64_iu8(true, af0, true, bf1, acc[0][1], false, false);
        acc[1][1] = __builtin_amdgcn_wmma_i32_16x16x64_iu8(true, af1, true, bf1, acc[1][1], false, false);
        acc[0][2] = __builtin_amdgcn_wmma_i32_16x16x64_iu8(true, af0, true, bf2, acc[0][2], false, false);
        acc[1][2] = __builtin_amdgcn_wmma_i32_16x16x64_iu8(true, af1, true, bf2, acc[1][2], false, false);
        acc[0][3] = __builtin_amdgcn_wmma_i32_16x16x64_iu8(true, af0, true, bf3, acc[0][3], false, false);
        acc[1][3] = __builtin_amdgcn_wmma_i32_16x16x64_iu8(true, af1, true, bf3, acc[1][3], false, false);

        wait_asynccnt0();
        __syncthreads();
    }

    const float inv127 = 1.0f / 127.0f;
    const int mshift = (lane >> 4) << 3;
#pragma unroll
    for (int ni = 0; ni < 4; ++ni) {
        const int n = col0 + ni * 16 + ml;
        const float rsn = wscale[n] * alpha[n] * inv127;
#pragma unroll
        for (int mi = 0; mi < 2; ++mi) {
            const int mbase = row0 + mi * 16 + mshift;
#pragma unroll
            for (int j = 0; j < 8; ++j) {
                const int m = mbase + j;
                Y[(size_t)m * N + n] = (float)acc[mi][ni][j] * rsn * ascale[m];
            }
        }
    }
}

extern "C" void kernel_launch(void* const* d_in, const int* in_sizes, int n_in,
                              void* d_out, int out_size, void* d_ws, size_t ws_size,
                              hipStream_t stream) {
    const float* x      = (const float*)d_in[0];
    const float* weight = (const float*)d_in[1];
    const float* alpha  = (const float*)d_in[2];
    float* y = (float*)d_out;

    const int DOUT = in_sizes[2];
    const int DIN  = in_sizes[1] / DOUT;
    const int M    = in_sizes[0] / DIN;
    const int K = DIN, N = DOUT;

    signed char* Aq = (signed char*)d_ws;
    signed char* Wq = Aq + (size_t)M * K;
    float* ascale   = (float*)(Wq + (size_t)N * K);
    float* wscale   = ascale + M;

    aquant_kernel<<<M, 256, 0, stream>>>(x, Aq, ascale, K);
    wquant_kernel<<<N, 256, 0, stream>>>(weight, Wq, wscale, K);

    dim3 grid(N / TILE_N, M / TILE_M);
    bitgemm_kernel<<<grid, 256, 0, stream>>>(Aq, Wq, ascale, wscale, alpha, y, M, N, K);
}
